// MultiHeadAttention_498216207009
// MI455X (gfx1250) — matrix operand order repaired and checked by reading, not compiled, not run on hardware
//
#include <hip/hip_runtime.h>
#include <hip/hip_bf16.h>

// MI455X / gfx1250 implementation. All GEMM work on v_wmma_f32_16x16x32_f16
// (f16 A/B, f32 accumulate). Workload is HBM-bound on the 1 GiB rel_pos_emb
// stream (read once, K-contiguous, prefetched); compute is ~47 GFLOP total.
// 8-wave (256-thread) workgroups with per-wave tiles to cut dispatch overhead.

typedef __attribute__((ext_vector_type(16))) _Float16 v16h;
typedef __attribute__((ext_vector_type(8)))  _Float16 v8h;
typedef __attribute__((ext_vector_type(8)))  float    v8f;

#define D_MODEL 1024
#define NHEADS  16
#define DK      64
#define BATCH   8
#define SEQ     512
#define MROWS   (BATCH * SEQ)   // 4096

#define WMMA_F16(a, b, c) \
  __builtin_amdgcn_wmma_f32_16x16x32_f16(false, (a), false, (b), (short)0, (c), false, false)

// ---------- fragment loaders (contiguous K order k = 16*(L>>4) + j, wave32) ----------

// A fragment, 16(M) x 32(K) f16, element A[m][k] = base[m*ld + k], f32 source.
// Lane L: M = L&15; halves j = K kb+j (j = 0..15),
// kb = (L>>4)*16.
static __device__ __forceinline__ v16h ld_a_f32(const float* __restrict__ base,
                                                int ld, int lane) {
  const int m  = lane & 15;
  const int kb = (lane >> 4) * 16;
  const float* p = base + (long)m * ld + kb;
  v16h a;
#pragma unroll
  for (int j = 0; j < 8; ++j) {
    a[j]     = (_Float16)p[j];
    a[j + 8] = (_Float16)p[j + 8];
  }
  return a;
}

static __device__ __forceinline__ v16h ld_a_f16(const _Float16* __restrict__ base,
                                                int ld, int lane) {
  const int m  = lane & 15;
  const int kb = (lane >> 4) * 16;
  const _Float16* p = base + (long)m * ld + kb;
  const v8h lo = *(const v8h*)(p);
  const v8h hi = *(const v8h*)(p + 8);
  v16h a;
#pragma unroll
  for (int j = 0; j < 8; ++j) { a[j] = lo[j]; a[j + 8] = hi[j]; }
  return a;
}

// B fragment, 32(K) x 16(N) f16, element B[k][n] = base[n*ld + k] (K-contiguous,
// i.e. loading X^T where X is row-major NxK). Lane L: N = L&15,
// halves j = K kb+j with kb = (L>>4)*16.
static __device__ __forceinline__ v16h ld_bT_f32(const float* __restrict__ base,
                                                 int ld, int lane) {
  const int n  = lane & 15;
  const int kb = (lane >> 4) * 16;
  const float* p = base + (long)n * ld + kb;
  v16h b;
#pragma unroll
  for (int j = 0; j < 16; ++j) b[j] = (_Float16)p[j];
  return b;
}

static __device__ __forceinline__ v16h ld_bT_f16(const _Float16* __restrict__ base,
                                                 int ld, int lane) {
  const int n  = lane & 15;
  const int kb = (lane >> 4) * 16;
  const _Float16* p = base + (long)n * ld + kb;
  const v8h lo = *(const v8h*)(p);
  const v8h hi = *(const v8h*)(p + 8);
  v16h b;
#pragma unroll
  for (int j = 0; j < 8; ++j) { b[j] = lo[j]; b[j + 8] = hi[j]; }
  return b;
}

// B fragment from row-major KxN source: B[k][n] = base[k*ld + n] (strided in K).
static __device__ __forceinline__ v16h ld_b_rm_f16(const _Float16* __restrict__ base,
                                                   int ld, int lane) {
  const int n  = lane & 15;
  const int kb = (lane >> 4) * 16;
  const _Float16* p = base + (long)kb * ld + n;
  v16h b;
#pragma unroll
  for (int j = 0; j < 16; ++j) b[j] = p[(long)j * ld];
  return b;
}

// ---------- K1: Y = X @ W^T + bias, write f16 in [B,H,L,dk] head layout ------
// 8 waves/block: wave w owns output row-tile blockIdx.y*8 + w.
__global__ void __launch_bounds__(256)
proj_kernel(const float* __restrict__ X, const float* __restrict__ W,
            const float* __restrict__ bias, _Float16* __restrict__ Yh) {
  const int lane = threadIdx.x & 31;
  const int wv   = threadIdx.x >> 5;
  const int nt = blockIdx.x * 16;                 // output column tile
  const int mt = (blockIdx.y * 8 + wv) * 16;      // output row tile
  v8f c = {};
#pragma unroll 4
  for (int k0 = 0; k0 < D_MODEL; k0 += 32) {
    v16h a = ld_a_f32(X + (long)mt * D_MODEL + k0, D_MODEL, lane);
    v16h b = ld_bT_f32(W + (long)nt * D_MODEL + k0, D_MODEL, lane);  // W^T
    c = WMMA_F16(a, b, c);
  }
  const int n = nt + (lane & 15);
  const int h = n >> 6, d = n & 63;
  const float bn = bias[n];
  const int mo = (lane >> 4) * 8;
#pragma unroll
  for (int i = 0; i < 8; ++i) {
    const int m = mt + mo + i;                    // row in [0, B*L)
    const int bi = m >> 9, l = m & 511;
    Yh[((((long)bi * NHEADS + h) * SEQ + l) * DK) + d] = (_Float16)(c[i] + bn);
  }
}

// ---------- K2: S = scale * qh @ kh^T  (f16 scores) --------------------------
// Block = (ltile, bh); wave w owns 4 r-tiles, A fragments loaded once per wave.
__global__ void __launch_bounds__(256)
qk_kernel(const _Float16* __restrict__ qh, const _Float16* __restrict__ kh,
          _Float16* __restrict__ S) {
  const int lane = threadIdx.x & 31;
  const int wv   = threadIdx.x >> 5;
  const int lt = blockIdx.x * 16;
  const int bh = blockIdx.y;
  const _Float16* qbase = qh + (long)bh * SEQ * DK + (long)lt * DK;
  const _Float16* kbase = kh + (long)bh * SEQ * DK;
  const v16h a0 = ld_a_f16(qbase + 0,  DK, lane);
  const v16h a1 = ld_a_f16(qbase + 32, DK, lane);
  const float scale = 0.125f;  // 1/sqrt(dk)
  const int nn = lane & 15;
  const int mo = (lane >> 4) * 8;
  _Float16* Sb = S + (long)bh * SEQ * SEQ;
#pragma unroll
  for (int j = 0; j < 4; ++j) {
    const int rt = (wv * 4 + j) * 16;
    v8f c = {};
    c = WMMA_F16(a0, ld_bT_f16(kbase + (long)rt * DK + 0,  DK, lane), c);
    c = WMMA_F16(a1, ld_bT_f16(kbase + (long)rt * DK + 32, DK, lane), c);
    const int r = rt + nn;
#pragma unroll
    for (int i = 0; i < 8; ++i) {
      const int l = lt + mo + i;
      Sb[(long)l * SEQ + r] = (_Float16)(c[i] * scale);
    }
  }
}

// ---------- K3: S += einsum('bhld,hlrd->bhlr') with batch packed as M --------
// rel[h,l,r,d] = rel_pos_emb[l, r, h*dk + d]; 1 GiB streamed once, K-contiguous.
// Block = (l, h); wave w owns 4 r-tiles; A (q rows for all 8 batches) loaded once.
__global__ void __launch_bounds__(256)
rel_kernel(const _Float16* __restrict__ qh, const float* __restrict__ relp,
           _Float16* __restrict__ S) {
  const int lane = threadIdx.x & 31;
  const int wv   = threadIdx.x >> 5;
  const int l = blockIdx.x;
  const int h = blockIdx.y;
  // A fragments: row M = batch index (rows 0..7 valid, 8..15 duplicates, discarded)
  const int m   = lane & 15;
  const int kb8 = (lane >> 4) * 16;
  const _Float16* qrow = qh + ((((long)(m & 7) * NHEADS + h) * SEQ + l) * DK) + kb8;
  v16h a0, a1;
  {
    const v8h l0 = *(const v8h*)(qrow);
    const v8h h0 = *(const v8h*)(qrow + 8);
    const v8h l1 = *(const v8h*)(qrow + 32);
    const v8h h1 = *(const v8h*)(qrow + 40);
#pragma unroll
    for (int j = 0; j < 8; ++j) {
      a0[j] = l0[j]; a0[j + 8] = h0[j];
      a1[j] = l1[j]; a1[j + 8] = h1[j];
    }
  }
  // B fragments: B[d][r] = rel_pos_emb[l, r, h*64 + d], d contiguous per lane.
  const int nn    = lane & 15;
  const int kb16  = (lane >> 4) * 16;
  const float* rel0 = relp + (((long)l * SEQ + nn) * D_MODEL) + h * DK + kb16;
#pragma unroll
  for (int j = 0; j < 4; ++j) {
    const int rt = (wv * 4 + j) * 16;
    const float* relb = rel0 + (long)rt * D_MODEL;
    if (j < 3) {  // prefetch next r-tile of this lane's stream (64 KB ahead)
      __builtin_prefetch(relb + 16L * D_MODEL, 0, 1);
      __builtin_prefetch(relb + 16L * D_MODEL + 32, 0, 1);
    }
    v16h b0, b1;
#pragma unroll
    for (int t = 0; t < 16; ++t) { b0[t] = (_Float16)relb[t]; b1[t] = (_Float16)relb[t + 32]; }
    v8f c = {};
    c = WMMA_F16(a0, b0, c);
    c = WMMA_F16(a1, b1, c);
    if (lane < 16) {                     // rows M=0..7 (b=i) live in lanes 0..15
      const int r = rt + lane;
#pragma unroll
      for (int i = 0; i < 8; ++i) {
        const long idx = ((((long)i * NHEADS + h) * SEQ + l) * SEQ) + r;
        S[idx] = (_Float16)((float)S[idx] + c[i]);
      }
    }
  }
}

// ---------- K4: mask + row softmax, in place (S -> P), one wave per row ------
__global__ void __launch_bounds__(256)
softmax_kernel(_Float16* __restrict__ S, const int* __restrict__ mask) {
  const int lane = threadIdx.x & 31;
  const int wv   = threadIdx.x >> 5;
  const long row = (long)blockIdx.x * 8 + wv;    // [0, B*H*L)
  const int l = (int)(row & (SEQ - 1));
  _Float16* Sr = S + row * SEQ;
  const int* Mr = mask + (long)l * SEQ;          // mask[0,0,l,r]
  float v[16];
  float mx = -3.0e38f;
#pragma unroll
  for (int j = 0; j < 16; ++j) {
    const int r = lane * 16 + j;
    float s = (float)Sr[r];
    s = (Mr[r] == 0) ? -1.0e9f : s;
    v[j] = s;
    mx = fmaxf(mx, s);
  }
#pragma unroll
  for (int o = 16; o > 0; o >>= 1) mx = fmaxf(mx, __shfl_xor(mx, o, 32));
  float sum = 0.0f;
#pragma unroll
  for (int j = 0; j < 16; ++j) { v[j] = __expf(v[j] - mx); sum += v[j]; }
#pragma unroll
  for (int o = 16; o > 0; o >>= 1) sum += __shfl_xor(sum, o, 32);
  const float inv = 1.0f / sum;
#pragma unroll
  for (int j = 0; j < 16; ++j) Sr[lane * 16 + j] = (_Float16)(v[j] * inv);
}

// ---------- K5: O[b,l,h*dk+d] = sum_r P[b,h,l,r] * vh[b,h,r,d] ---------------
// Block = (lt-pair, bh); wave w: dt = w&3, lt = pair*2 + (w>>2).
__global__ void __launch_bounds__(256)
pv_kernel(const _Float16* __restrict__ P, const _Float16* __restrict__ vh,
          _Float16* __restrict__ O) {
  const int lane = threadIdx.x & 31;
  const int wv   = threadIdx.x >> 5;
  const int dt = (wv & 3) * 16;
  const int lt = (blockIdx.x * 2 + (wv >> 2)) * 16;
  const int bh = blockIdx.y;
  const int b = bh >> 4, h = bh & 15;
  const _Float16* Pb = P + (long)bh * SEQ * SEQ + (long)lt * SEQ;
  const _Float16* Vb = vh + (long)bh * SEQ * DK + dt;
  v8f c = {};
  for (int r0 = 0; r0 < SEQ; r0 += 32) {
    v16h a  = ld_a_f16(Pb + r0, SEQ, lane);                 // A[l][r]
    v16h bf = ld_b_rm_f16(Vb + (long)r0 * DK, DK, lane);    // B[r][d]
    c = WMMA_F16(a, bf, c);
  }
  const int d = dt + (lane & 15);
  const int mo = (lane >> 4) * 8;
#pragma unroll
  for (int i = 0; i < 8; ++i) {
    const int l = lt + mo + i;
    O[((long)b * SEQ + l) * D_MODEL + h * DK + d] = (_Float16)c[i];
  }
}

// ---------- K6: out = O @ Wo^T + bo  (fp32 output) ---------------------------
__global__ void __launch_bounds__(256)
oproj_kernel(const _Float16* __restrict__ O, const float* __restrict__ Wo,
             const float* __restrict__ bo, float* __restrict__ out) {
  const int lane = threadIdx.x & 31;
  const int wv   = threadIdx.x >> 5;
  const int nt = blockIdx.x * 16;
  const int mt = (blockIdx.y * 8 + wv) * 16;
  v8f c = {};
#pragma unroll 4
  for (int k0 = 0; k0 < D_MODEL; k0 += 32) {
    v16h a = ld_a_f16(O + (long)mt * D_MODEL + k0, D_MODEL, lane);
    v16h b = ld_bT_f32(Wo + (long)nt * D_MODEL + k0, D_MODEL, lane);
    c = WMMA_F16(a, b, c);
  }
  const int n = nt + (lane & 15);
  const float bn = bo[n];
  const int mo = (lane >> 4) * 8;
#pragma unroll
  for (int i = 0; i < 8; ++i) {
    const int m = mt + mo + i;
    out[(long)m * D_MODEL + n] = c[i] + bn;
  }
}

extern "C" void kernel_launch(void* const* d_in, const int* in_sizes, int n_in,
                              void* d_out, int out_size, void* d_ws, size_t ws_size,
                              hipStream_t stream) {
  (void)in_sizes; (void)n_in; (void)out_size; (void)ws_size;
  const float* q    = (const float*)d_in[0];
  const float* k    = (const float*)d_in[1];
  const float* v    = (const float*)d_in[2];
  const float* rel  = (const float*)d_in[3];
  const int*   mask = (const int*)d_in[4];
  const float* Wq   = (const float*)d_in[5];
  const float* bq   = (const float*)d_in[6];
  const float* Wk   = (const float*)d_in[7];
  const float* bk   = (const float*)d_in[8];
  const float* Wv   = (const float*)d_in[9];
  const float* bv   = (const float*)d_in[10];
  const float* Wo   = (const float*)d_in[11];
  const float* bo   = (const float*)d_in[12];
  float* out = (float*)d_out;

  // Workspace layout (96 MiB total): qh/kh/vh (8 MiB each, f16 head layout),
  // S/P (64 MiB f16 scores, softmaxed in place), O (8 MiB f16).
  char* ws = (char*)d_ws;
  const size_t headBytes = (size_t)BATCH * NHEADS * SEQ * DK * sizeof(_Float16);
  _Float16* qh = (_Float16*)ws;  ws += headBytes;
  _Float16* kh = (_Float16*)ws;  ws += headBytes;
  _Float16* vh = (_Float16*)ws;  ws += headBytes;
  _Float16* S  = (_Float16*)ws;  ws += (size_t)BATCH * NHEADS * SEQ * SEQ * sizeof(_Float16);
  _Float16* O  = (_Float16*)ws;

  const dim3 blk(256);

  const dim3 gProj(D_MODEL / 16, MROWS / 16 / 8);          // (64, 32)
  proj_kernel<<<gProj, blk, 0, stream>>>(q, Wq, bq, qh);
  proj_kernel<<<gProj, blk, 0, stream>>>(k, Wk, bk, kh);
  proj_kernel<<<gProj, blk, 0, stream>>>(v, Wv, bv, vh);

  const dim3 gQK(SEQ / 16, BATCH * NHEADS);                // (32, 128)
  qk_kernel<<<gQK, blk, 0, stream>>>(qh, kh, S);

  const dim3 gRel(SEQ, NHEADS);                            // (512, 16)
  rel_kernel<<<gRel, blk, 0, stream>>>(qh, rel, S);

  softmax_kernel<<<dim3(BATCH * NHEADS * SEQ / 8), blk, 0, stream>>>(S, mask);

  const dim3 gPV(SEQ / 32, BATCH * NHEADS);                // (16, 128)
  pv_kernel<<<gPV, blk, 0, stream>>>(S, vh, O);

  const dim3 gO(D_MODEL / 16, MROWS / 16 / 8);             // (64, 32)
  oproj_kernel<<<gO, blk, 0, stream>>>(O, Wo, bo, out);
}
